// Tiny_ConvGRUCell_27754078667596
// MI455X (gfx1250) — compile-verified, matrix operand order checked by reading, not run on hardware
//
#include <hip/hip_runtime.h>

typedef __attribute__((ext_vector_type(16))) _Float16 v16h;
typedef __attribute__((ext_vector_type(8)))  float    v8f;

#define HW      65536      // H*W
#define PLANE   2097152    // 32*HW (one batch image, all channels)
#define NTILES  32768      // (8*65536)/16
#define HALF_N  16777216   // elements in y (= elements in h_t)

struct Params {
    const float *x, *h;
    const float *wxz, *bxz, *whz, *bhz;
    const float *wxr, *bxr, *whr, *bhr;
    const float *wc,  *bc,  *wu,  *bu;
    const float *wo,  *bo;
    float *y, *ht;
};

__device__ __forceinline__ float fast_sigmoid(float x) {
    float e = __builtin_amdgcn_exp2f(x * -1.44269504f);     // v_exp_f32
    return __builtin_amdgcn_rcpf(1.0f + e);                 // v_rcp_f32
}
__device__ __forceinline__ float fast_tanh(float x) {
    float e = __builtin_amdgcn_exp2f(x * -2.88539008f);
    return __builtin_amdgcn_rcpf(1.0f + e) * 2.0f - 1.0f;
}

#define WMMA_F16(A, B, C) \
    __builtin_amdgcn_wmma_f32_16x16x32_f16(false, (A), false, (B), (short)0, (C), false, false)

__global__ __launch_bounds__(256) void gru1x1_wmma(Params p) {
    // ---- LDS: weights (row stride 33 for conflict-free fragment reads) + combined biases
    __shared__ float wlds[7 * 1056];   // 7 x (32 rows * 33)
    __shared__ float blds[4 * 32];     // bz, br, bc+bu, bo

    const int tid = threadIdx.x;
    {
        const float* wsrc[7] = { p.wxz, p.whz, p.wxr, p.whr, p.wc, p.wu, p.wo };
        #pragma unroll
        for (int m = 0; m < 7; ++m)
            for (int idx = tid; idx < 1024; idx += 256)
                wlds[m * 1056 + (idx >> 5) * 33 + (idx & 31)] = wsrc[m][idx];
        if (tid < 32) {
            blds[      tid] = p.bxz[tid] + p.bhz[tid];
            blds[ 32 + tid] = p.bxr[tid] + p.bhr[tid];
            blds[ 64 + tid] = p.bc[tid]  + p.bu[tid];
            blds[ 96 + tid] = p.bo[tid];
        }
    }
    __syncthreads();

    const int lane = tid & 31;
    const int Lm   = lane & 15;     // A-matrix row / B-matrix column (pixel)
    const int hi   = lane >> 4;     // lane group: K offset +8

    // ---- Build weight A-fragments (16x32 f16): wf[mat][out-half]
    // A layout: lane L<16 holds M=L, K = {2j+t (j<4), 16+2(j-4)+t (j>=4)}; lane L+16: K += 8
    v16h wf[7][2];
    #pragma unroll
    for (int m = 0; m < 7; ++m)
        #pragma unroll
        for (int f = 0; f < 2; ++f)
            #pragma unroll
            for (int e = 0; e < 16; ++e) {
                int j = e >> 1, t = e & 1;
                int kk = ((j < 4) ? 2 * j : 2 * j + 8) + 8 * hi + t;
                wf[m][f][e] = (_Float16)wlds[m * 1056 + (16 * f + Lm) * 33 + kk];
            }

    const int waveId = tid >> 5;
    const int nwaves = gridDim.x * 8;

    for (int tile = blockIdx.x * 8 + waveId; tile < NTILES; tile += nwaves) {
        const int b   = tile >> 12;               // 4096 tiles per batch plane
        const int hw0 = (tile & 4095) << 4;
        const size_t base = (size_t)b * PLANE + hw0 + Lm + (size_t)hi * (8 * HW);
        const float* xb = p.x + base;
        const float* hb = p.h + base;

        // ---- Prefetch next tile while this one computes (pure streaming kernel)
        {
            int nt = tile + nwaves;
            if (nt < NTILES) {
                int nb = nt >> 12, nhw = (nt & 4095) << 4;
                const float* nx = p.x + (size_t)nb * PLANE + nhw + Lm + (size_t)hi * (8 * HW);
                const float* nh = p.h + (size_t)nb * PLANE + nhw + Lm + (size_t)hi * (8 * HW);
                #pragma unroll
                for (int e = 0; e < 16; ++e) {
                    int k = (e < 8) ? e : e + 8;
                    __builtin_prefetch(nx + (size_t)k * HW, 0, 3);   // global_prefetch
                    __builtin_prefetch(nh + (size_t)k * HW, 0, 3);
                }
            }
        }

        // ---- Load x/h B-fragments (32ch x 16px, f16); channel offsets are immediates
        v16h bx, bh;
        #pragma unroll
        for (int e = 0; e < 16; ++e) {
            int k = (e < 8) ? e : e + 8;           // + 8*hi folded into base
            bx[e] = (_Float16)xb[(size_t)k * HW];
            bh[e] = (_Float16)hb[(size_t)k * HW];
        }

        // ---- Gates z, r: D = Wx*x + Wh*h + bias (bias as WMMA C operand)
        v8f zacc[2], racc[2];
        #pragma unroll
        for (int f = 0; f < 2; ++f) {
            const int bo = 16 * f + 8 * hi;
            v8f cz, cr;
            #pragma unroll
            for (int k = 0; k < 8; ++k) { cz[k] = blds[bo + k]; cr[k] = blds[32 + bo + k]; }
            zacc[f] = WMMA_F16(wf[0][f], bx, cz);
            zacc[f] = WMMA_F16(wf[1][f], bh, zacc[f]);
            racc[f] = WMMA_F16(wf[2][f], bx, cr);
            racc[f] = WMMA_F16(wf[3][f], bh, racc[f]);
        }

        // ---- Elementwise: sigmoid; build (r .* h) B-fragment (pure per-lane repack)
        float zv[16];
        v16h brh;
        #pragma unroll
        for (int e = 0; e < 16; ++e) {
            int f = e >> 3, k = e & 7;
            zv[e]  = fast_sigmoid(zacc[f][k]);
            float r = fast_sigmoid(racc[f][k]);
            brh[e] = (_Float16)r * bh[e];
        }

        // ---- Candidate: tanh(Wc*x + Wu*(r.*h) + bias)
        v8f cacc[2];
        #pragma unroll
        for (int f = 0; f < 2; ++f) {
            const int bo = 16 * f + 8 * hi;
            v8f cc;
            #pragma unroll
            for (int k = 0; k < 8; ++k) cc[k] = blds[64 + bo + k];
            cacc[f] = WMMA_F16(wf[4][f], bx, cc);
            cacc[f] = WMMA_F16(wf[5][f], brh, cacc[f]);
        }

        // ---- h_t = (1-z)*h + z*hhat : store + repack as B-fragment for output conv
        float* yb  = p.y  + base;
        float* hob = p.ht + base;
        v16h bht;
        #pragma unroll
        for (int e = 0; e < 16; ++e) {
            int f = e >> 3, k = e & 7;
            int kk = (e < 8) ? e : e + 8;
            float hhat = fast_tanh(cacc[f][k]);
            float hval = (float)bh[e];
            float htv  = (1.0f - zv[e]) * hval + zv[e] * hhat;
            hob[(size_t)kk * HW] = htv;
            bht[e] = (_Float16)htv;
        }

        // ---- Output conv: y = Wo*h_t + bo
        #pragma unroll
        for (int f = 0; f < 2; ++f) {
            const int bo = 16 * f + 8 * hi;
            v8f co;
            #pragma unroll
            for (int k = 0; k < 8; ++k) co[k] = blds[96 + bo + k];
            v8f yacc = WMMA_F16(wf[6][f], bht, co);
            #pragma unroll
            for (int k = 0; k < 8; ++k) {
                int kk = k + 16 * f;               // + 8*hi folded into base
                yb[(size_t)kk * HW] = yacc[k];
            }
        }
    }
}

extern "C" void kernel_launch(void* const* d_in, const int* in_sizes, int n_in,
                              void* d_out, int out_size, void* d_ws, size_t ws_size,
                              hipStream_t stream) {
    (void)in_sizes; (void)n_in; (void)d_ws; (void)ws_size; (void)out_size;
    Params p;
    p.x   = (const float*)d_in[0];
    p.h   = (const float*)d_in[1];
    p.wxz = (const float*)d_in[2];  p.bxz = (const float*)d_in[3];
    p.whz = (const float*)d_in[4];  p.bhz = (const float*)d_in[5];
    p.wxr = (const float*)d_in[6];  p.bxr = (const float*)d_in[7];
    p.whr = (const float*)d_in[8];  p.bhr = (const float*)d_in[9];
    p.wc  = (const float*)d_in[10]; p.bc  = (const float*)d_in[11];
    p.wu  = (const float*)d_in[12]; p.bu  = (const float*)d_in[13];
    p.wo  = (const float*)d_in[14]; p.bo  = (const float*)d_in[15];
    p.y   = (float*)d_out;
    p.ht  = p.y + HALF_N;
    hipLaunchKernelGGL(gru1x1_wmma, dim3(512), dim3(256), 0, stream, p);
}
